// GATWithPooling_28836410425433
// MI455X (gfx1250) — matrix operand order repaired and checked by reading, not compiled, not run on hardware
//
#include <hip/hip_runtime.h>
#include <hip/hip_bf16.h>
#include <math.h>
#include <string.h>

typedef __attribute__((ext_vector_type(16))) __bf16 v16bf;
typedef __attribute__((ext_vector_type(8)))  float  v8f;

#define HEADS 4
#define HID 128
#define F 512            /* HEADS*HID */
#define NEG_SLOPE 0.2f

/* ---------- helpers ---------- */
__device__ __forceinline__ unsigned short f2bf(float f) {
    unsigned u = __float_as_uint(f);
    unsigned r = u + 0x7FFFu + ((u >> 16) & 1u);   // round-to-nearest-even
    return (unsigned short)(r >> 16);
}
__device__ __forceinline__ unsigned okey(float f) {  // monotonic uint key
    unsigned u = __float_as_uint(f);
    return (u & 0x80000000u) ? ~u : (u | 0x80000000u);
}
__device__ __forceinline__ float keyf(unsigned k) {  // inverse
    return (k & 0x80000000u) ? __uint_as_float(k ^ 0x80000000u)
                             : __uint_as_float(~k);
}

/* ---------- conversions ---------- */
__global__ void k_f2bf(const float* __restrict__ s, unsigned short* __restrict__ d, int n) {
    int t = blockIdx.x * blockDim.x + threadIdx.x;
    if (t < n) d[t] = f2bf(s[t]);
}
/* W [K x Nw] fp32 row-major -> Wt [Nw x K] bf16 row-major */
__global__ void k_w_to_bf16t(const float* __restrict__ W, unsigned short* __restrict__ Wt,
                             int K, int Nw) {
    int t = blockIdx.x * blockDim.x + threadIdx.x;
    if (t >= K * Nw) return;
    int k = t / Nw, n = t % Nw;
    Wt[(size_t)n * K + k] = f2bf(W[t]);
}

/* ---------- bf16 WMMA GEMM: C[M,Nn] = A[M,K] * Bt[Nn,K]^T ----------
   One wave computes a 16(M) x 64(N) tile: the A fragment is loaded once per
   k-step and reused across 4 WMMAs (2 A-loads + 8 B-loads per 4 WMMAs). */
__global__ void k_gemm_bf16(const unsigned short* __restrict__ A,
                            const unsigned short* __restrict__ Bt,
                            float* __restrict__ C,
                            int M, int Nn, int K, int tilesN4, int totalTiles) {
    int wid = blockIdx.x * (blockDim.x >> 5) + (threadIdx.x >> 5);
    if (wid >= totalTiles) return;          // wave-uniform exit: EXEC all-1 for WMMA
    int lane = threadIdx.x & 31;
    int m0 = (wid / tilesN4) * 16;
    int n0 = (wid % tilesN4) * 64;
    int hiA = (lane >> 4) << 4;             // A: 0 or 16 (K chunks of 16)
    int hiB = (lane >> 4) << 4;             // B: 0 or 16 (K chunks of 16)
    const unsigned short* aP = A  + (size_t)(m0 + (lane & 15)) * K;
    const unsigned short* bP = Bt + (size_t)(n0 + (lane & 15)) * K;
    v8f acc0 = {}, acc1 = {}, acc2 = {}, acc3 = {};
    union Frag { uint4 u[2]; v16bf v; };
    for (int kk = 0; kk < K; kk += 32) {
        Frag a, b0, b1, b2, b3;
        a.u[0]  = *(const uint4*)(aP + kk + hiA);        // K kk+{0..7} / {16..23}
        a.u[1]  = *(const uint4*)(aP + kk + 8 + hiA);    // K kk+{8..15} / {24..31}
        b0.u[0] = *(const uint4*)(bP + (size_t)0 * 16 * K + kk + hiB);
        b0.u[1] = *(const uint4*)(bP + (size_t)0 * 16 * K + kk + 8 + hiB);
        b1.u[0] = *(const uint4*)(bP + (size_t)1 * 16 * K + kk + hiB);
        b1.u[1] = *(const uint4*)(bP + (size_t)1 * 16 * K + kk + 8 + hiB);
        b2.u[0] = *(const uint4*)(bP + (size_t)2 * 16 * K + kk + hiB);
        b2.u[1] = *(const uint4*)(bP + (size_t)2 * 16 * K + kk + 8 + hiB);
        b3.u[0] = *(const uint4*)(bP + (size_t)3 * 16 * K + kk + hiB);
        b3.u[1] = *(const uint4*)(bP + (size_t)3 * 16 * K + kk + 8 + hiB);
        acc0 = __builtin_amdgcn_wmma_f32_16x16x32_bf16(false, a.v, false, b0.v, (short)0, acc0, false, false);
        acc1 = __builtin_amdgcn_wmma_f32_16x16x32_bf16(false, a.v, false, b1.v, (short)0, acc1, false, false);
        acc2 = __builtin_amdgcn_wmma_f32_16x16x32_bf16(false, a.v, false, b2.v, (short)0, acc2, false, false);
        acc3 = __builtin_amdgcn_wmma_f32_16x16x32_bf16(false, a.v, false, b3.v, (short)0, acc3, false, false);
    }
    int rbase = m0 + ((lane >> 4) << 3);
    int cbase = n0 + (lane & 15);
#pragma unroll
    for (int r = 0; r < 8; ++r) {
        int row = rbase + r;
        if (row < M) {
            float* cp = C + (size_t)row * Nn + cbase;
            cp[0]  = acc0[r];
            cp[16] = acc1[r];
            cp[32] = acc2[r];
            cp[48] = acc3[r];
        }
    }
}

/* ---------- per-node attention coefficients: asrc/adst [n,4] ---------- */
__global__ void k_node_attn(const float* __restrict__ h,
                            const float* __restrict__ a_src, const float* __restrict__ a_dst,
                            float* __restrict__ asrc, float* __restrict__ adst, int n) {
    int w = blockIdx.x * (blockDim.x >> 5) + (threadIdx.x >> 5);
    if (w >= n * HEADS) return;
    int lane = threadIdx.x & 31;
    int node = w >> 2, head = w & 3;
    const float* hp = h + (size_t)node * F + head * HID;
    const float* sp = a_src + head * HID;
    const float* dp = a_dst + head * HID;
    float s1 = 0.f, s2 = 0.f;
    for (int j = lane; j < HID; j += 32) { float hv = hp[j]; s1 += hv * sp[j]; s2 += hv * dp[j]; }
    for (int off = 16; off; off >>= 1) { s1 += __shfl_down(s1, off, 32); s2 += __shfl_down(s2, off, 32); }
    if (lane == 0) { asrc[w] = s1; adst[w] = s2; }
}

__global__ void k_fill_u32(unsigned* p, unsigned v, int n) {
    int t = blockIdx.x * blockDim.x + threadIdx.x;
    if (t < n) p[t] = v;
}

/* ---------- edge pass A: segment max (float as ordered uint) ---------- */
__global__ void k_edge_max(const int* __restrict__ src, const int* __restrict__ dst,
                           const float* __restrict__ asrc, const float* __restrict__ adst,
                           unsigned* __restrict__ mx, int E, int nloops) {
    int t = blockIdx.x * blockDim.x + threadIdx.x;
    if (t >= E + nloops) return;
    int s, d;
    if (t < E) { s = src[t]; d = dst[t]; if (s < 0 || d < 0) return; }
    else       { s = d = t - E; }
    float4 as = ((const float4*)asrc)[s];
    float4 ad = ((const float4*)adst)[d];
    float l[4] = {as.x + ad.x, as.y + ad.y, as.z + ad.z, as.w + ad.w};
#pragma unroll
    for (int hh = 0; hh < 4; ++hh) {
        float lv = l[hh] > 0.f ? l[hh] : NEG_SLOPE * l[hh];
        atomicMax(&mx[d * 4 + hh], okey(lv));
    }
}

/* ---------- edge pass B: e = exp(logit - max), den += e ---------- */
__global__ void k_edge_exp(const int* __restrict__ src, const int* __restrict__ dst,
                           const float* __restrict__ asrc, const float* __restrict__ adst,
                           const unsigned* __restrict__ mx, float* __restrict__ ebuf,
                           float* __restrict__ den, int E, int nloops) {
    int t = blockIdx.x * blockDim.x + threadIdx.x;
    if (t >= E + nloops) return;
    int s, d;
    if (t < E) { s = src[t]; d = dst[t]; if (s < 0 || d < 0) return; }
    else       { s = d = t - E; }
    float4 as = ((const float4*)asrc)[s];
    float4 ad = ((const float4*)adst)[d];
    float l[4] = {as.x + ad.x, as.y + ad.y, as.z + ad.z, as.w + ad.w};
#pragma unroll
    for (int hh = 0; hh < 4; ++hh) {
        float lv = l[hh] > 0.f ? l[hh] : NEG_SLOPE * l[hh];
        float e = expf(lv - keyf(mx[d * 4 + hh]));
        ebuf[t * 4 + hh] = e;
        atomicAdd(&den[d * 4 + hh], e);
    }
}

/* ---------- edge pass C: acc[dst] += alpha * h[src] (float4 per thread) ---------- */
__global__ void k_edge_scatter(const int* __restrict__ src, const int* __restrict__ dst,
                               const float* __restrict__ ebuf, const float* __restrict__ den,
                               const float* __restrict__ h, float* __restrict__ acc,
                               int E, int nloops) {
    long long t = (long long)blockIdx.x * blockDim.x + threadIdx.x;
    int e = (int)(t >> 7);
    if (e >= E + nloops) return;
    int chunk = (int)(t & 127);
    int s, d;
    if (e < E) { s = src[e]; d = dst[e]; if (s < 0 || d < 0) return; }
    else       { s = d = e - E; }
    int head = chunk >> 5;
    float alpha = ebuf[e * 4 + head] / den[d * 4 + head];
    int cc = chunk << 2;
    float4 hv = *(const float4*)(h + (size_t)s * F + cc);
    float* ap = acc + (size_t)d * F + cc;
    atomicAdd(ap + 0, alpha * hv.x);
    atomicAdd(ap + 1, alpha * hv.y);
    atomicAdd(ap + 2, alpha * hv.z);
    atomicAdd(ap + 3, alpha * hv.w);
}

/* ---------- bias + ELU, in place ---------- */
__global__ void k_bias_elu(float* acc, const float* __restrict__ bias, int rows) {
    int t = blockIdx.x * blockDim.x + threadIdx.x;
    if (t >= rows * F) return;
    float v = acc[t] + bias[t & (F - 1)];
    acc[t] = v > 0.f ? v : (expf(v) - 1.f);
}

/* ---------- TopK pooling pieces ---------- */
__global__ void k_pw_prep(const float* __restrict__ pw, float* invnorm) {
    __shared__ float sh[512];
    int t = threadIdx.x;
    float v = pw[t];
    sh[t] = v * v; __syncthreads();
    for (int s = 256; s; s >>= 1) { if (t < s) sh[t] += sh[t + s]; __syncthreads(); }
    if (t == 0) invnorm[0] = rsqrtf(sh[0]);
}

__global__ void k_score(const float* __restrict__ hx, const float* __restrict__ pw,
                        const float* __restrict__ invnorm, float* __restrict__ score, int n) {
    int w = blockIdx.x * (blockDim.x >> 5) + (threadIdx.x >> 5);
    if (w >= n) return;
    int lane = threadIdx.x & 31;
    const float* hp = hx + (size_t)w * F;
    float s = 0.f;
    for (int j = lane; j < F; j += 32) s += hp[j] * pw[j];
    for (int off = 16; off; off >>= 1) s += __shfl_down(s, off, 32);
    if (lane == 0) score[w] = tanhf(s * invnorm[0]);
}

/* exact deterministic top-k select: remap[n] (-1 = dropped), perm[k] (kept ids, idx order) */
__global__ void k_select(const float* __restrict__ score, int n, int k,
                         int* __restrict__ remap, int* __restrict__ perm) {
    __shared__ unsigned scnt[1024];
    __shared__ int sscan[1024];
    __shared__ unsigned s_lo, s_hi, s_gt;
    __shared__ int s_base, s_eqbase;
    int t = threadIdx.x;
    if (t == 0) { s_lo = 0u; s_hi = 0xFFFFFFFFu; }
    __syncthreads();
    for (int it = 0; it < 34; ++it) {               // binary search kth-largest key
        unsigned lo = s_lo, hi = s_hi;
        if (lo >= hi) break;
        unsigned mid = lo + ((hi - lo) >> 1) + 1u;
        unsigned c = 0;
        for (int i = t; i < n; i += 1024) c += (okey(score[i]) >= mid) ? 1u : 0u;
        scnt[t] = c; __syncthreads();
        for (int s = 512; s; s >>= 1) { if (t < s) scnt[t] += scnt[t + s]; __syncthreads(); }
        if (t == 0) { if (scnt[0] >= (unsigned)k) s_lo = mid; else s_hi = mid - 1u; }
        __syncthreads();
    }
    unsigned T = s_lo;
    {   unsigned c = 0;                              // count strictly greater
        for (int i = t; i < n; i += 1024) c += (okey(score[i]) > T) ? 1u : 0u;
        scnt[t] = c; __syncthreads();
        for (int s = 512; s; s >>= 1) { if (t < s) scnt[t] += scnt[t + s]; __syncthreads(); }
        if (t == 0) { s_gt = scnt[0]; s_base = 0; s_eqbase = 0; }
        __syncthreads();
    }
    int need = k - (int)s_gt;                        // ties to keep (index order)
    for (int base = 0; base < n; base += 1024) {
        int i = base + t, gt = 0, eq = 0;
        if (i < n) { unsigned key = okey(score[i]); gt = key > T; eq = (key == T); }
        sscan[t] = eq; __syncthreads();
        for (int off = 1; off < 1024; off <<= 1) {
            int x = (t >= off) ? sscan[t - off] : 0; __syncthreads();
            sscan[t] += x; __syncthreads();
        }
        int eqEx = sscan[t] - eq, eqTot = sscan[1023]; __syncthreads();
        int keep = (i < n) && (gt || (eq && (s_eqbase + eqEx) < need));
        sscan[t] = keep; __syncthreads();
        for (int off = 1; off < 1024; off <<= 1) {
            int x = (t >= off) ? sscan[t - off] : 0; __syncthreads();
            sscan[t] += x; __syncthreads();
        }
        int kEx = sscan[t] - keep, kTot = sscan[1023]; __syncthreads();
        if (i < n) {
            if (keep) { int pos = s_base + kEx; remap[i] = pos; perm[pos] = i; }
            else remap[i] = -1;
        }
        __syncthreads();
        if (t == 0) { s_base += kTot; s_eqbase += eqTot; }
        __syncthreads();
    }
}

__global__ void k_pool_gather_bf16(const float* __restrict__ hx, const float* __restrict__ score,
                                   const int* __restrict__ perm, unsigned short* __restrict__ out,
                                   int k) {
    int t = blockIdx.x * blockDim.x + threadIdx.x;
    if (t >= k * F) return;
    int i = t >> 9, c = t & 511;
    int node = perm[i];
    out[t] = f2bf(hx[(size_t)node * F + c] * score[node]);
}

__global__ void k_edge_remap(const int* __restrict__ src, const int* __restrict__ dst,
                             const int* __restrict__ remap,
                             int* __restrict__ src2, int* __restrict__ dst2, int E) {
    int t = blockIdx.x * blockDim.x + threadIdx.x;
    if (t >= E) return;
    src2[t] = remap[src[t]];
    dst2[t] = remap[dst[t]];
}

/* ---------- final: masked mean + linear ---------- */
__global__ void k_mean_accum(const float* __restrict__ hx, const float* __restrict__ score,
                             const int* __restrict__ perm, float* __restrict__ g, int k) {
    int t = blockIdx.x * blockDim.x + threadIdx.x;
    if (t >= k * F) return;
    int i = t >> 9, c = t & 511;
    int node = perm[i];
    atomicAdd(&g[c], hx[(size_t)node * F + c] * score[node]);
}

__global__ void k_final(const float* __restrict__ g, const float* __restrict__ Wl,
                        const float* __restrict__ bl, float* __restrict__ out,
                        int k2, int OUTD) {
    int o = threadIdx.x;
    if (o >= OUTD) return;
    float inv = 1.f / (float)k2;
    float acc = bl[o];
    for (int c = 0; c < F; ++c) acc += g[c] * inv * Wl[c * OUTD + o];
    out[o] = acc;
}

/* ================= host ================= */
static inline unsigned okey_host(float f) {
    unsigned u; memcpy(&u, &f, 4);
    return (u & 0x80000000u) ? ~u : (u | 0x80000000u);
}
static inline unsigned blks(long long n, int bs) { return (unsigned)((n + bs - 1) / bs); }

extern "C" void kernel_launch(void* const* d_in, const int* in_sizes, int n_in,
                              void* d_out, int out_size, void* d_ws, size_t ws_size,
                              hipStream_t stream) {
    const float* x      = (const float*)d_in[0];
    const int*   eidx   = (const int*)d_in[1];
    const float* W1     = (const float*)d_in[3];
    const float* a_src1 = (const float*)d_in[4];
    const float* a_dst1 = (const float*)d_in[5];
    const float* b1     = (const float*)d_in[6];
    const float* pw1    = (const float*)d_in[7];
    const float* W2     = (const float*)d_in[8];
    const float* a_src2 = (const float*)d_in[9];
    const float* a_dst2 = (const float*)d_in[10];
    const float* b2     = (const float*)d_in[11];
    const float* pw2    = (const float*)d_in[12];
    const float* Wl     = (const float*)d_in[13];
    const float* bl     = (const float*)d_in[14];
    float* out = (float*)d_out;

    const int IN = 64;
    const int N  = in_sizes[0] / IN;       // 20000
    const int E  = in_sizes[1] / 2;        // 200000
    const int K1 = (N + 1) / 2;            // 10000
    const int K2 = (K1 + 1) / 2;           // 5000
    const int OUTD = out_size;             // 10
    const int* srcI = eidx;
    const int* dstI = eidx + E;

    /* ws bump allocator (256B aligned) */
    char* wsb = (char*)d_ws; size_t off = 0;
    auto alloc = [&](size_t bytes) -> void* {
        void* p = wsb + off; off += (bytes + 255) & ~(size_t)255; return p;
    };
    unsigned short* xbf  = (unsigned short*)alloc((size_t)N * IN * 2);
    unsigned short* Wt1  = (unsigned short*)alloc((size_t)F * IN * 2);
    unsigned short* Wt2  = (unsigned short*)alloc((size_t)F * F * 2);
    float* h    = (float*)alloc((size_t)N * F * 4);   // h1, reused as h2
    float* acc  = (float*)alloc((size_t)N * F * 4);   // acc1/hx1, reused acc2/hx2
    float* asrc = (float*)alloc((size_t)N * HEADS * 4);
    float* adst = (float*)alloc((size_t)N * HEADS * 4);
    unsigned* mx = (unsigned*)alloc((size_t)N * HEADS * 4);
    float* den  = (float*)alloc((size_t)N * HEADS * 4);
    float* ebuf = (float*)alloc((size_t)(E + N) * HEADS * 4);
    float* score = (float*)alloc((size_t)N * 4);
    int* remap  = (int*)alloc((size_t)N * 4);
    int* perm   = (int*)alloc((size_t)K1 * 4);
    int* src2   = (int*)alloc((size_t)E * 4);
    int* dst2   = (int*)alloc((size_t)E * 4);
    unsigned short* x2bf = (unsigned short*)alloc((size_t)K1 * F * 2);
    float* invn = (float*)alloc(256);
    float* g    = (float*)alloc((size_t)F * 4);
    if (off > ws_size) return;  // insufficient scratch; bail deterministically

    const int BS = 256;
    const unsigned NEGKEY = okey_host(-1e30f);
    const int tilesN4 = F / 64;            // 8 groups of 4 N-tiles

    /* conversions */
    k_f2bf<<<blks((long long)N * IN, BS), BS, 0, stream>>>(x, xbf, N * IN);
    k_w_to_bf16t<<<blks((long long)IN * F, BS), BS, 0, stream>>>(W1, Wt1, IN, F);
    k_w_to_bf16t<<<blks((long long)F * F, BS), BS, 0, stream>>>(W2, Wt2, F, F);

    /* ===== layer 1 ===== */
    {
        int tiles = ((N + 15) / 16) * tilesN4;
        k_gemm_bf16<<<blks(tiles, 8), BS, 0, stream>>>(xbf, Wt1, h, N, F, IN, tilesN4, tiles);
        k_node_attn<<<blks((long long)N * HEADS, 8), BS, 0, stream>>>(h, a_src1, a_dst1, asrc, adst, N);
        k_fill_u32<<<blks((long long)N * HEADS, BS), BS, 0, stream>>>(mx, NEGKEY, N * HEADS);
        k_edge_max<<<blks((long long)E + N, BS), BS, 0, stream>>>(srcI, dstI, asrc, adst, mx, E, N);
        hipMemsetAsync(den, 0, (size_t)N * HEADS * 4, stream);
        k_edge_exp<<<blks((long long)E + N, BS), BS, 0, stream>>>(srcI, dstI, asrc, adst, mx, ebuf, den, E, N);
        hipMemsetAsync(acc, 0, (size_t)N * F * 4, stream);
        k_edge_scatter<<<blks((long long)(E + N) * 128, BS), BS, 0, stream>>>(srcI, dstI, ebuf, den, h, acc, E, N);
        k_bias_elu<<<blks((long long)N * F, BS), BS, 0, stream>>>(acc, b1, N);
        /* pool 1 */
        k_pw_prep<<<1, 512, 0, stream>>>(pw1, invn);
        k_score<<<blks(N, 8), BS, 0, stream>>>(acc, pw1, invn, score, N);
        k_select<<<1, 1024, 0, stream>>>(score, N, K1, remap, perm);
        k_pool_gather_bf16<<<blks((long long)K1 * F, BS), BS, 0, stream>>>(acc, score, perm, x2bf, K1);
        k_edge_remap<<<blks(E, BS), BS, 0, stream>>>(srcI, dstI, remap, src2, dst2, E);
    }

    /* ===== layer 2 ===== */
    {
        int tiles = ((K1 + 15) / 16) * tilesN4;
        k_gemm_bf16<<<blks(tiles, 8), BS, 0, stream>>>(x2bf, Wt2, h, K1, F, F, tilesN4, tiles);
        k_node_attn<<<blks((long long)K1 * HEADS, 8), BS, 0, stream>>>(h, a_src2, a_dst2, asrc, adst, K1);
        k_fill_u32<<<blks((long long)K1 * HEADS, BS), BS, 0, stream>>>(mx, NEGKEY, K1 * HEADS);
        k_edge_max<<<blks((long long)E + K1, BS), BS, 0, stream>>>(src2, dst2, asrc, adst, mx, E, K1);
        hipMemsetAsync(den, 0, (size_t)K1 * HEADS * 4, stream);
        k_edge_exp<<<blks((long long)E + K1, BS), BS, 0, stream>>>(src2, dst2, asrc, adst, mx, ebuf, den, E, K1);
        hipMemsetAsync(acc, 0, (size_t)K1 * F * 4, stream);
        k_edge_scatter<<<blks((long long)(E + K1) * 128, BS), BS, 0, stream>>>(src2, dst2, ebuf, den, h, acc, E, K1);
        k_bias_elu<<<blks((long long)K1 * F, BS), BS, 0, stream>>>(acc, b2, K1);
        /* pool 2 */
        k_pw_prep<<<1, 512, 0, stream>>>(pw2, invn);
        k_score<<<blks(K1, 8), BS, 0, stream>>>(acc, pw2, invn, score, K1);
        k_select<<<1, 1024, 0, stream>>>(score, K1, K2, remap, perm);
    }

    /* ===== mean pool + linear ===== */
    hipMemsetAsync(g, 0, (size_t)F * 4, stream);
    k_mean_accum<<<blks((long long)K2 * F, BS), BS, 0, stream>>>(acc, score, perm, g, K2);
    k_final<<<1, 64, 0, stream>>>(g, Wl, bl, out, K2, OUTD);
}
